// Baseline_Q7_test_72688026517894
// MI455X (gfx1250) — matrix operand order repaired and checked by reading, not compiled, not run on hardware
//
#include <hip/hip_runtime.h>

typedef __attribute__((ext_vector_type(16))) _Float16 v16h;
typedef __attribute__((ext_vector_type(8)))  float    v8f;

#define S_DIM 32
#define H_DIM 128
#define KPAD  64
#define NEG_SLOPE 0.01f
#define LN_EPS 1e-5f
#define WAVES 4

__device__ __forceinline__ void wave_lds_fence() {
    asm volatile("s_wait_dscnt 0" ::: "memory");
}
__device__ __forceinline__ void wave_async_fence() {
    asm volatile("s_wait_asynccnt 0" ::: "memory");
}

__device__ __forceinline__ float lrelu(float x) {
    return x >= 0.f ? x : NEG_SLOPE * x;
}

// A-fragment element i (0..15) -> k offset within a 32-wide K chunk
// (16-bit A 16x32 layout, ISA 7.12.2): v = i/2, half = i%2,
// k = (v>=4)*16 + (v&3)*2 + (laneHi*8) + half
__device__ __forceinline__ v16h load_a_frag(const _Float16* rowBase, int c, int hh) {
    v16h a;
#pragma unroll
    for (int i = 0; i < 16; i++) {
        int v = i >> 1;
        int k = c * 32 + ((v & 4) ? 16 : 0) + ((v & 3) << 1) + (hh << 3) + (i & 1);
        a[i] = rowBase[k];
    }
    return a;
}

// ---------------------------------------------------------------------------
// Fused message-passing round: async gather -> (33x128 GEMM+LN+lrelu) ->
// (128x32 GEMM+LN+lrelu) -> atomic scatter.  One wave = 16 edges per tile.
// ---------------------------------------------------------------------------
__global__ __launch_bounds__(128) void round_kernel(
    const _Float16* __restrict__ stateH, float* __restrict__ stateNext,
    const int* __restrict__ node_from, const int* __restrict__ node_to,
    const float* __restrict__ edge_c,
    const float* __restrict__ W1, const float* __restrict__ b1,
    const float* __restrict__ g1, const float* __restrict__ be1,
    const float* __restrict__ W2, const float* __restrict__ b2,
    const float* __restrict__ g2, const float* __restrict__ be2,
    int E, int numTiles)
{
    // Pre-swizzled weight fragments: one v16h per lane per fragment, contiguous.
    __shared__ _Float16 sW1[16 * 512];          // 16 frags (2 Kchunks x 8 Ntiles)
    __shared__ _Float16 sW2[8 * 512];           // 8 frags  (4 Kchunks x 2 Ntiles)
    __shared__ _Float16 sInp[WAVES][16 * KPAD]; // per-wave 16x64 f16 input tile
    __shared__ _Float16 sH[WAVES][16 * H_DIM];  // per-wave 16x128 f16 hidden tile

    const int tid  = threadIdx.x;
    const int wave = tid >> 5;
    const int lane = tid & 31;
    const int m    = lane & 15;     // row within 16-edge tile / N index
    const int hh   = lane >> 4;     // lane-half

    // Build W1 B-fragments (K padded 33 -> 64 with zeros).
    for (int idx = tid; idx < 16 * 512; idx += blockDim.x) {
        int f = idx >> 9, r = idx & 511;
        int ln = r >> 4, s = r & 15;
        int c = f >> 3, t = f & 7;
        int n = t * 16 + (ln & 15);
        int k = c * 32 + ((s & 8) ? 16 : 0) + ((ln >> 4) << 3) + (s & 7);   // B 32x16: k = 16*(s>=8) + 8*laneHi + (s&7)
        float val = (k < S_DIM + 1) ? W1[k * H_DIM + n] : 0.f;
        sW1[idx] = (_Float16)val;
    }
    // Build W2 B-fragments (128x32, exact).
    for (int idx = tid; idx < 8 * 512; idx += blockDim.x) {
        int f = idx >> 9, r = idx & 511;
        int ln = r >> 4, s = r & 15;
        int c = f >> 1, t = f & 1;
        int n = t * 16 + (ln & 15);
        int k = c * 32 + ((s & 8) ? 16 : 0) + ((ln >> 4) << 3) + (s & 7);
        sW2[idx] = (_Float16)W2[k * S_DIM + n];
    }
    __syncthreads();

    _Float16* sInpW = sInp[wave];
    _Float16* sHW   = sH[wave];

    // W2 fragments live in registers for the whole kernel (8 x v16h = 64 VGPRs).
    v16h bw2[8];
#pragma unroll
    for (int f = 0; f < 8; f++) bw2[f] = *(const v16h*)&sW2[f * 512 + lane * 16];

    // Zero the K padding (33..63) once; only k<=32 is rewritten per tile.
    if (hh == 0) {
        for (int j = S_DIM + 1; j < KPAD; j++) sInpW[m * KPAD + j] = (_Float16)0.f;
    }

    // Per-lane column constants (col = 16*t + n).
    float b1v[8], g1v[8], be1v[8];
#pragma unroll
    for (int t = 0; t < 8; t++) {
        int col = t * 16 + m;
        b1v[t] = b1[col]; g1v[t] = g1[col]; be1v[t] = be1[col];
    }
    float b2v[2], g2v[2], be2v[2];
#pragma unroll
    for (int t = 0; t < 2; t++) {
        int col = t * 16 + m;
        b2v[t] = b2[col]; g2v[t] = g2[col]; be2v[t] = be2[col];
    }

    const int stride = gridDim.x * WAVES;
    for (int tile = blockIdx.x * WAVES + wave; tile < numTiles; tile += stride) {
        const int eBase = tile * 16;

        // ---- Stage inputs: async copy 32B (16 halves) per lane, global->LDS ----
        {
            int e = eBase + m; if (e >= E) e = E - 1;
            int src = node_from[e];
            unsigned long long gAddr =
                (unsigned long long)(const void*)(stateH + (size_t)src * S_DIM + hh * 16);
            unsigned int ldsOff =
                (unsigned int)(size_t)(const void*)(sInpW + m * KPAD + hh * 16);
            asm volatile(
                "global_load_async_to_lds_b128 %0, %1, off\n\t"
                "global_load_async_to_lds_b128 %0, %1, off offset:16"
                :: "v"(ldsOff), "v"(gAddr) : "memory");
            if (hh) sInpW[m * KPAD + S_DIM] = (_Float16)edge_c[e];
        }
        wave_async_fence();
        wave_lds_fence();

        // ---- Layer 1: 16x64 (f16) @ 64x128 -> 16x128 f32, 16 WMMAs ----
        v8f acc[8] = {};
#pragma unroll
        for (int c = 0; c < 2; c++) {
            v16h a = load_a_frag(sInpW + m * KPAD, c, hh);
#pragma unroll
            for (int t = 0; t < 8; t++) {
                v16h b = *(const v16h*)&sW1[(c * 8 + t) * 512 + lane * 16];
                acc[t] = __builtin_amdgcn_wmma_f32_16x16x32_f16(
                    false, a, false, b, (short)0, acc[t], false, false);
            }
        }

        // ---- + b1, LayerNorm over 128, lrelu, store f16 to sH ----
#pragma unroll
        for (int t = 0; t < 8; t++)
#pragma unroll
            for (int v = 0; v < 8; v++) acc[t][v] += b1v[t];

        float sm[8], sq[8];
#pragma unroll
        for (int v = 0; v < 8; v++) {
            float x = 0.f, x2 = 0.f;
#pragma unroll
            for (int t = 0; t < 8; t++) { float y = acc[t][v]; x += y; x2 += y * y; }
            sm[v] = x; sq[v] = x2;
        }
#pragma unroll
        for (int mask = 1; mask < 16; mask <<= 1)
#pragma unroll
            for (int v = 0; v < 8; v++) {
                sm[v] += __shfl_xor(sm[v], mask, 32);
                sq[v] += __shfl_xor(sq[v], mask, 32);
            }
        float mean[8], inv[8];
#pragma unroll
        for (int v = 0; v < 8; v++) {
            mean[v] = sm[v] * (1.f / H_DIM);
            float var = sq[v] * (1.f / H_DIM) - mean[v] * mean[v];
            inv[v] = rsqrtf(var + LN_EPS);
        }
#pragma unroll
        for (int t = 0; t < 8; t++)
#pragma unroll
            for (int v = 0; v < 8; v++) {
                float y = (acc[t][v] - mean[v]) * inv[v] * g1v[t] + be1v[t];
                y = lrelu(y);
                sHW[(v + 8 * hh) * H_DIM + t * 16 + m] = (_Float16)y;
            }
        wave_lds_fence();

        // ---- Layer 2: 16x128 @ 128x32 -> 16x32 f32, 8 WMMAs ----
        v8f acc2[2] = {};
#pragma unroll
        for (int c = 0; c < 4; c++) {
            v16h a = load_a_frag(sHW + m * H_DIM, c, hh);
#pragma unroll
            for (int t = 0; t < 2; t++) {
                acc2[t] = __builtin_amdgcn_wmma_f32_16x16x32_f16(
                    false, a, false, bw2[c * 2 + t], (short)0, acc2[t], false, false);
            }
        }

        // ---- + b2, LayerNorm over 32, lrelu, atomic scatter ----
#pragma unroll
        for (int t = 0; t < 2; t++)
#pragma unroll
            for (int v = 0; v < 8; v++) acc2[t][v] += b2v[t];

        float sm2[8], sq2[8];
#pragma unroll
        for (int v = 0; v < 8; v++) {
            float a0 = acc2[0][v], a1 = acc2[1][v];
            sm2[v] = a0 + a1; sq2[v] = a0 * a0 + a1 * a1;
        }
#pragma unroll
        for (int mask = 1; mask < 16; mask <<= 1)
#pragma unroll
            for (int v = 0; v < 8; v++) {
                sm2[v] += __shfl_xor(sm2[v], mask, 32);
                sq2[v] += __shfl_xor(sq2[v], mask, 32);
            }
        int dstv[8];
#pragma unroll
        for (int v = 0; v < 8; v++) {
            int e2 = eBase + v + 8 * hh; if (e2 >= E) e2 = E - 1;
            dstv[v] = node_to[e2];
        }
#pragma unroll
        for (int v = 0; v < 8; v++) {
            float mean2 = sm2[v] * (1.f / S_DIM);
            float var2  = sq2[v] * (1.f / S_DIM) - mean2 * mean2;
            float inv2  = rsqrtf(var2 + LN_EPS);
            int e2 = eBase + v + 8 * hh;
            if (e2 < E) {
#pragma unroll
                for (int t = 0; t < 2; t++) {
                    float y = (acc2[t][v] - mean2) * inv2 * g2v[t] + be2v[t];
                    y = lrelu(y);
                    atomicAdd(&stateNext[(size_t)dstv[v] * S_DIM + t * 16 + m], y);
                }
            }
        }
    }
}

// ---------------------------------------------------------------------------
__global__ void zero_kernel(float* p, long n) {
    long i = (long)blockIdx.x * blockDim.x + threadIdx.x;
    if (i < n) p[i] = 0.f;
}

__global__ void zero_h_kernel(_Float16* p, long n) {
    long i = (long)blockIdx.x * blockDim.x + threadIdx.x;
    if (i < n) p[i] = (_Float16)0.f;
}

__global__ void conv_h_kernel(const float* __restrict__ src,
                              _Float16* __restrict__ dst, long n) {
    long i = (long)blockIdx.x * blockDim.x + threadIdx.x;
    if (i < n) dst[i] = (_Float16)src[i];
}

__global__ void pool_kernel(const float* __restrict__ state,
                            const int* __restrict__ ngi,
                            float* __restrict__ gs, long total) {
    long i = (long)blockIdx.x * blockDim.x + threadIdx.x;
    if (i < total) {
        int node = (int)(i >> 5);
        int col  = (int)(i & 31);
        atomicAdd(&gs[(size_t)ngi[node] * S_DIM + col], state[i]);
    }
}

__device__ __forceinline__ float block_reduce_128(float v, float* red, int tid) {
    red[tid] = v; __syncthreads();
#pragma unroll
    for (int off = 64; off > 0; off >>= 1) {
        if (tid < off) red[tid] += red[tid + off];
        __syncthreads();
    }
    float r = red[0]; __syncthreads();
    return r;
}

__global__ __launch_bounds__(128) void out_kernel(
    const float* __restrict__ gs,
    const float* __restrict__ Wo1, const float* __restrict__ bo1,
    const float* __restrict__ go,  const float* __restrict__ beo,
    const float* __restrict__ Wo2, const float* __restrict__ bo2,
    float* __restrict__ out, int G)
{
    __shared__ float red[128];
    int g = blockIdx.x;
    if (g >= G) return;
    int j = threadIdx.x;
    const float* row = gs + (size_t)g * S_DIM;

    float acc = bo1[j];
#pragma unroll
    for (int k = 0; k < S_DIM; k++) acc += row[k] * Wo1[k * H_DIM + j];

    float sum = block_reduce_128(acc, red, j);
    float sq  = block_reduce_128(acc * acc, red, j);
    float mean = sum * (1.f / H_DIM);
    float var  = sq * (1.f / H_DIM) - mean * mean;
    float h = (acc - mean) * rsqrtf(var + LN_EPS) * go[j] + beo[j];
    h = lrelu(h);

    float ev0 = block_reduce_128(h * Wo2[j * 2 + 0], red, j) + bo2[0];
    float ev1 = block_reduce_128(h * Wo2[j * 2 + 1], red, j) + bo2[1];
    if (j == 0) {
        out[(size_t)g * 2 + 0] = ev0;
        out[(size_t)g * 2 + 1] = (ev1 > 20.f) ? ev1 : log1pf(expf(ev1));
    }
}

// ---------------------------------------------------------------------------
extern "C" void kernel_launch(void* const* d_in, const int* in_sizes, int n_in,
                              void* d_out, int out_size, void* d_ws, size_t ws_size,
                              hipStream_t stream) {
    const int*   node_from = (const int*)d_in[0];
    const int*   node_to   = (const int*)d_in[1];
    const float* edge_c    = (const float*)d_in[2];
    const int*   ngi       = (const int*)d_in[3];
    const float* W1  = (const float*)d_in[6];
    const float* b1  = (const float*)d_in[7];
    const float* g1  = (const float*)d_in[8];
    const float* be1 = (const float*)d_in[9];
    const float* W2  = (const float*)d_in[10];
    const float* b2  = (const float*)d_in[11];
    const float* g2  = (const float*)d_in[12];
    const float* be2 = (const float*)d_in[13];
    const float* Wo1 = (const float*)d_in[14];
    const float* bo1 = (const float*)d_in[15];
    const float* go  = (const float*)d_in[16];
    const float* beo = (const float*)d_in[17];
    const float* Wo2 = (const float*)d_in[18];
    const float* bo2 = (const float*)d_in[19];

    const int E = in_sizes[0];
    const int N = in_sizes[3];
    const int G = out_size / 2;

    float*    stateA = (float*)d_ws;
    float*    stateB = stateA + (size_t)N * S_DIM;
    float*    gs     = stateB + (size_t)N * S_DIM;
    _Float16* stateH = (_Float16*)(gs + (size_t)G * S_DIM);

    long nS = (long)N * S_DIM;
    zero_kernel<<<(int)((nS + 255) / 256), 256, 0, stream>>>(stateA, nS);
    zero_h_kernel<<<(int)((nS + 255) / 256), 256, 0, stream>>>(stateH, nS);

    int numTiles = (E + 15) / 16;
    int blocks = (numTiles + WAVES - 1) / WAVES;
    if (blocks > 1024) blocks = 1024;

    float* cur = stateA;
    float* nxt = stateB;
    for (int r = 0; r < 5; r++) {
        hipMemcpyAsync(nxt, cur, nS * sizeof(float), hipMemcpyDeviceToDevice, stream);
        round_kernel<<<blocks, 128, 0, stream>>>(
            stateH, nxt, node_from, node_to, edge_c,
            W1, b1, g1, be1, W2, b2, g2, be2, E, numTiles);
        conv_h_kernel<<<(int)((nS + 255) / 256), 256, 0, stream>>>(nxt, stateH, nS);
        float* t = cur; cur = nxt; nxt = t;
    }

    long gsz = (long)G * S_DIM;
    zero_kernel<<<(int)((gsz + 255) / 256), 256, 0, stream>>>(gs, gsz);
    pool_kernel<<<(int)((nS + 255) / 256), 256, 0, stream>>>(cur, ngi, gs, nS);
    out_kernel<<<G, 128, 0, stream>>>(gs, Wo1, bo1, go, beo, Wo2, bo2, (float*)d_out, G);
}
